// SAGE_27410481283336
// MI455X (gfx1250) — matrix operand order repaired and checked by reading, not compiled, not run on hardware
//
#include <hip/hip_runtime.h>
#include <hip/hip_bf16.h>

typedef __attribute__((ext_vector_type(16))) __bf16 v16bf;
typedef __attribute__((ext_vector_type(8)))  __bf16 v8bf;
typedef __attribute__((ext_vector_type(8)))  float  v8f;

#define N0v 600000
#define N1v 200000
#define N2v 50000
#define N3v 10000
#define FANOUTv 15
#define DIM_IN 128
#define DIM_HID 256
#define DIM_OUT 64

// ---------------- fp32 -> bf16 conversion ----------------
__global__ void cvt_f32_bf16(const float* __restrict__ in, __bf16* __restrict__ out, int n) {
    int i = blockIdx.x * blockDim.x + threadIdx.x;
    if (i < n) out[i] = (__bf16)in[i];
}

// ---------------- in-degree count ----------------
__global__ void deg_count(const int* __restrict__ dst, float* __restrict__ deg, int E) {
    int i = blockIdx.x * blockDim.x + threadIdx.x;
    if (i < E) atomicAdd(&deg[dst[i]], 1.0f);
}

// ---------------- edge scatter-add of raw features (aggregate-first) ----------------
// grid.x covers edges, grid.y covers feature chunks of 4
__global__ void scatter_add(const __bf16* __restrict__ feat, const int* __restrict__ src,
                            const int* __restrict__ dst, float* __restrict__ agg,
                            int E, int K) {
    int e = blockIdx.x * blockDim.x + threadIdx.x;
    if (e >= E) return;
    int c = blockIdx.y << 2;
    int s = src[e], d = dst[e];
    const __bf16* p = feat + (size_t)s * K + c;
    float* q = agg + (size_t)d * K + c;
    atomicAdd(q + 0, (float)p[0]);
    atomicAdd(q + 1, (float)p[1]);
    atomicAdd(q + 2, (float)p[2]);
    atomicAdd(q + 3, (float)p[3]);
}

// ---------------- fused SAGE layer: dual WMMA GEMM + epilogue ----------------
// out[d, n] = maybe_relu( feat[d,:] @ Wself[n,:] + bias[n] + (agg[d,:] @ Wneigh[n,:]) / deg[d] )
// Each wave computes a 16(M) x 64(N) tile: 4 col-tiles x 2 accumulator sets.
__global__ void __launch_bounds__(256)
sage_fused(const __bf16* __restrict__ feat,   // [M,K] bf16 (dst-node features)
           const float*  __restrict__ agg,    // [M,K] f32 (summed neighbor features)
           const float*  __restrict__ deg,    // [M]
           const __bf16* __restrict__ Wself,  // [N,K] bf16
           const __bf16* __restrict__ Wneigh, // [N,K] bf16
           const float*  __restrict__ bias,   // [N]
           float* __restrict__ out_f32,       // [M,N] or null
           __bf16* __restrict__ out_bf16,     // [M,N] or null
           int M, int N, int K, int relu) {
    int wave = blockIdx.x * (blockDim.x >> 5) + (threadIdx.x >> 5);
    int lane = threadIdx.x & 31;
    int colGroups = N >> 6;                 // tiles of 64 cols
    int rowTiles  = M >> 4;
    if (wave >= rowTiles * colGroups) return;   // wave-uniform: EXEC stays all-1s
    int rt = wave / colGroups;
    int cg = wave % colGroups;
    int m0 = rt << 4;
    int n0 = cg << 6;
    int half = lane >> 4;                   // 0: lanes 0-15, 1: lanes 16-31
    int r    = lane & 15;

    v8f accS[4] = {};
    v8f accN[4] = {};

    const __bf16* pa = feat + (size_t)(m0 + r) * K;
    const float*  pn = agg  + (size_t)(m0 + r) * K;

    for (int k0 = 0; k0 < K; k0 += 32) {
        // A fragment (self path), bf16 16x32: lane r holds row r;
        // half=0 -> K k0..k0+15; half=1 -> +16
        int kb = k0 + half * 16;
        v16bf aS;
        {
            v8bf lo = *(const v8bf*)(pa + kb);
            v8bf hi = *(const v8bf*)(pa + kb + 8);
#pragma unroll
            for (int j = 0; j < 8; ++j) { aS[j] = lo[j]; aS[8 + j] = hi[j]; }
        }
        // A fragment (neighbor path): f32 -> bf16 on the fly
        v16bf aN;
        {
#pragma unroll
            for (int j = 0; j < 8; ++j) {
                aN[j]     = (__bf16)pn[kb + j];
                aN[8 + j] = (__bf16)pn[kb + 8 + j];
            }
        }
        int koff = k0 + half * 16;          // B: lanes 0-15 hold K k0..k0+15, lanes 16-31 +16
#pragma unroll
        for (int t = 0; t < 4; ++t) {
            int col = n0 + t * 16 + r;      // B column n = W row n (contiguous K)
            v16bf bS = *(const v16bf*)(Wself  + (size_t)col * K + koff);
            v16bf bN = *(const v16bf*)(Wneigh + (size_t)col * K + koff);
            accS[t] = __builtin_amdgcn_wmma_f32_16x16x32_bf16(
                false, aS, false, bS, (short)0, accS[t], false, false);
            accN[t] = __builtin_amdgcn_wmma_f32_16x16x32_bf16(
                false, aN, false, bN, (short)0, accN[t], false, false);
        }
    }

    // epilogue: C element i of this lane -> row m0 + half*8 + i, col = n0 + t*16 + r
    float invd[8];
#pragma unroll
    for (int i = 0; i < 8; ++i) invd[i] = 1.0f / deg[m0 + half * 8 + i];

#pragma unroll
    for (int t = 0; t < 4; ++t) {
        int col = n0 + t * 16 + r;
        float bv = bias[col];
#pragma unroll
        for (int i = 0; i < 8; ++i) {
            int row = m0 + half * 8 + i;
            float o = accS[t][i] + bv + accN[t][i] * invd[i];
            if (relu) o = fmaxf(o, 0.0f);
            if (out_f32)  out_f32[(size_t)row * N + col] = o;
            if (out_bf16) out_bf16[(size_t)row * N + col] = (__bf16)o;
        }
    }
}

extern "C" void kernel_launch(void* const* d_in, const int* in_sizes, int n_in,
                              void* d_out, int out_size, void* d_ws, size_t ws_size,
                              hipStream_t stream) {
    (void)in_sizes; (void)n_in; (void)out_size; (void)ws_size;
    const float* x    = (const float*)d_in[0];
    const int*   src0 = (const int*)d_in[1];
    const int*   dst0 = (const int*)d_in[2];
    const int*   src1 = (const int*)d_in[3];
    const int*   dst1 = (const int*)d_in[4];
    const int*   src2 = (const int*)d_in[5];
    const int*   dst2 = (const int*)d_in[6];
    const float* Wn0  = (const float*)d_in[7];
    const float* Ws0  = (const float*)d_in[8];
    const float* b0   = (const float*)d_in[9];
    const float* Wn1  = (const float*)d_in[10];
    const float* Ws1  = (const float*)d_in[11];
    const float* b1   = (const float*)d_in[12];
    const float* Wn2  = (const float*)d_in[13];
    const float* Ws2  = (const float*)d_in[14];
    const float* b2   = (const float*)d_in[15];

    const int E0 = N1v * FANOUTv, E1 = N2v * FANOUTv, E2 = N3v * FANOUTv;

    // ---- workspace carve-up (~385 MB) ----
    char* ws = (char*)d_ws;
    size_t off = 0;
    auto carve = [&](size_t bytes) -> char* {
        char* p = ws + off;
        off = (off + bytes + 255) & ~(size_t)255;
        return p;
    };
    __bf16* xb   = (__bf16*)carve((size_t)N0v * DIM_IN * 2);
    __bf16* Wn0b = (__bf16*)carve((size_t)DIM_HID * DIM_IN * 2);
    __bf16* Ws0b = (__bf16*)carve((size_t)DIM_HID * DIM_IN * 2);
    __bf16* Wn1b = (__bf16*)carve((size_t)DIM_HID * DIM_HID * 2);
    __bf16* Ws1b = (__bf16*)carve((size_t)DIM_HID * DIM_HID * 2);
    __bf16* Wn2b = (__bf16*)carve((size_t)DIM_OUT * DIM_HID * 2);
    __bf16* Ws2b = (__bf16*)carve((size_t)DIM_OUT * DIM_HID * 2);
    __bf16* h1   = (__bf16*)carve((size_t)N1v * DIM_HID * 2);
    __bf16* h2   = (__bf16*)carve((size_t)N2v * DIM_HID * 2);
    float*  agg  = (float*)carve((size_t)N1v * DIM_IN * 4);   // reused (largest of 3 layers)
    float*  deg  = (float*)carve((size_t)N1v * 4);            // reused

    const int T = 256;
    auto cdiv = [](long long a, long long b) { return (unsigned)((a + b - 1) / b); };

    // ---- precision conversions ----
    cvt_f32_bf16<<<cdiv((long long)N0v * DIM_IN, T), T, 0, stream>>>(x, xb, N0v * DIM_IN);
    cvt_f32_bf16<<<cdiv(DIM_HID * DIM_IN, T),  T, 0, stream>>>(Wn0, Wn0b, DIM_HID * DIM_IN);
    cvt_f32_bf16<<<cdiv(DIM_HID * DIM_IN, T),  T, 0, stream>>>(Ws0, Ws0b, DIM_HID * DIM_IN);
    cvt_f32_bf16<<<cdiv(DIM_HID * DIM_HID, T), T, 0, stream>>>(Wn1, Wn1b, DIM_HID * DIM_HID);
    cvt_f32_bf16<<<cdiv(DIM_HID * DIM_HID, T), T, 0, stream>>>(Ws1, Ws1b, DIM_HID * DIM_HID);
    cvt_f32_bf16<<<cdiv(DIM_OUT * DIM_HID, T), T, 0, stream>>>(Wn2, Wn2b, DIM_OUT * DIM_HID);
    cvt_f32_bf16<<<cdiv(DIM_OUT * DIM_HID, T), T, 0, stream>>>(Ws2, Ws2b, DIM_OUT * DIM_HID);

    // ---- layer 0: x(600k,128) -> h1(200k,256), relu ----
    hipMemsetAsync(deg, 0, (size_t)N1v * 4, stream);
    hipMemsetAsync(agg, 0, (size_t)N1v * DIM_IN * 4, stream);
    deg_count<<<cdiv(E0, T), T, 0, stream>>>(dst0, deg, E0);
    scatter_add<<<dim3(cdiv(E0, T), DIM_IN / 4), T, 0, stream>>>(xb, src0, dst0, agg, E0, DIM_IN);
    {
        unsigned waves = (N1v / 16) * (DIM_HID / 64);
        sage_fused<<<cdiv(waves, 8), 256, 0, stream>>>(
            xb, agg, deg, Ws0b, Wn0b, b0, nullptr, h1, N1v, DIM_HID, DIM_IN, 1);
    }

    // ---- layer 1: h1(200k,256) -> h2(50k,256), relu ----
    hipMemsetAsync(deg, 0, (size_t)N2v * 4, stream);
    hipMemsetAsync(agg, 0, (size_t)N2v * DIM_HID * 4, stream);
    deg_count<<<cdiv(E1, T), T, 0, stream>>>(dst1, deg, E1);
    scatter_add<<<dim3(cdiv(E1, T), DIM_HID / 4), T, 0, stream>>>(h1, src1, dst1, agg, E1, DIM_HID);
    {
        unsigned waves = (N2v / 16) * (DIM_HID / 64);
        sage_fused<<<cdiv(waves, 8), 256, 0, stream>>>(
            h1, agg, deg, Ws1b, Wn1b, b1, nullptr, h2, N2v, DIM_HID, DIM_HID, 1);
    }

    // ---- layer 2: h2(50k,256) -> out(10k,64), no relu, fp32 ----
    hipMemsetAsync(deg, 0, (size_t)N3v * 4, stream);
    hipMemsetAsync(agg, 0, (size_t)N3v * DIM_HID * 4, stream);
    deg_count<<<cdiv(E2, T), T, 0, stream>>>(dst2, deg, E2);
    scatter_add<<<dim3(cdiv(E2, T), DIM_HID / 4), T, 0, stream>>>(h2, src2, dst2, agg, E2, DIM_HID);
    {
        unsigned waves = (N3v / 16) * (DIM_OUT / 64);
        sage_fused<<<cdiv(waves, 8), 256, 0, stream>>>(
            h2, agg, deg, Ws2b, Wn2b, b2, (float*)d_out, nullptr, N3v, DIM_OUT, DIM_HID, 0);
    }
}
